// SelfConditionedConformerEncoder_8022998909709
// MI455X (gfx1250) — matrix operand order repaired and checked by reading, not compiled, not run on hardware
//
#include <hip/hip_runtime.h>
#include <hip/hip_bf16.h>
#include <math.h>

// ---------------------------------------------------------------------------
// SelfConditionedConformerEncoder for MI455X (gfx1250, wave32, WMMA).
// All GEMMs run on v_wmma_f32_16x16x32_bf16 (fp32 accumulate).  Weights are
// converted fp32->bf16 AND transposed to [N,K] once per launch (tiled LDS
// transpose), so every weight GEMM stages B with pure b128 copies (no
// transposed b16 scatter in the hot loop).  Activations on the A side are
// fp32 -> bf16 during LDS staging (global_load_b128 + v_cvt_pk_bf16_f32 +
// ds_store_b128).  128x128x32 block tiles, 8 waves (4x2), each wave 32x64
// (2x4 16x16 accs) -> 8 WMMA per k-step.  LDS double buffered, 80B row
// stride (16B-aligned ds_load_b128, conflict-free bank pattern),
// global_prefetch_b8 for the k+2 tile.  N-bounds guarding is a template
// parameter so only the attnV GEMM (N=64 < 128) pays for it.
// ---------------------------------------------------------------------------

#define LNUM 6
#define BBAT 8
#define NSEQ 512
#define DMOD 512
#define HHEADS 8
#define DHEAD 64
#define FFDIM 2048
#define KCONV 31
#define VVOC 128
#define MTOK (BBAT*NSEQ)   // 4096

typedef __attribute__((ext_vector_type(16))) __bf16 v16bf;
typedef __attribute__((ext_vector_type(8)))  float  v8f;

#define TBM 128
#define TBN 128
#define TBK 32
#define LDSS (TBK + 8)   // 40 bf16 = 80B row stride: 16B aligned, conflict-free

// ------------------------------ wave helpers -------------------------------
__device__ __forceinline__ float wred_sum(float v) {
    #pragma unroll
    for (int o = 16; o; o >>= 1) v += __shfl_xor(v, o, 32);
    return v;
}
__device__ __forceinline__ float wred_max(float v) {
    #pragma unroll
    for (int o = 16; o; o >>= 1) v = fmaxf(v, __shfl_xor(v, o, 32));
    return v;
}
__device__ __forceinline__ float silu_f(float x) { return x / (1.f + expf(-x)); }
__device__ __forceinline__ float sigm_f(float x) { return 1.f / (1.f + expf(-x)); }

// ---------- fp32 [L,K,N] -> bf16 [L,N,K] transpose (weights, once) ---------
// 32x32 tiles via LDS (stride 33 to dodge bank conflicts); both global sides
// coalesced.  grid = (K/32, N/32, L), block = 256 (32x8).
__global__ __launch_bounds__(256)
void cvtT_kernel(const float* __restrict__ src, __bf16* __restrict__ dst, int K, int N)
{
    __shared__ float tile[32][33];
    const int k0 = blockIdx.x * 32, n0 = blockIdx.y * 32;
    const float* s = src + (size_t)blockIdx.z * K * N;
    __bf16*      d = dst + (size_t)blockIdx.z * K * N;
    const int tx = threadIdx.x & 31, ty = threadIdx.x >> 5;
    #pragma unroll
    for (int r = 0; r < 32; r += 8)
        tile[ty + r][tx] = s[(size_t)(k0 + ty + r) * N + n0 + tx];
    __syncthreads();
    #pragma unroll
    for (int r = 0; r < 32; r += 8)
        d[(size_t)(n0 + ty + r) * K + k0 + tx] = (__bf16)tile[tx][ty + r];
}

// ------------------------------ WMMA GEMM ----------------------------------
// C[m,n] = epilogue( sc * sum_k A[m,k]*B[k,n] + bias[n] )
// A: fp32 (converted to bf16 while staging).  B: bf16 in memory.
// EPI 0: linear (+bias)   EPI 1: SiLU(linear)   EPI 2: Res + resScale*linear
// BT=true: B stored [N,K] row-major (all weight GEMMs + Q@K^T).
// NGUARD: N-dimension bounds checks (only the attnV GEMM needs them).
// Batched via grid.z; C offset = (z/cZdiv)*cS1 + (z%cZdiv)*cS2.
template <int EPI, bool BT, bool NGUARD>
__global__ __launch_bounds__(256)
void gemm_wmma(const float* __restrict__ A, long long sAz,
               const __bf16* __restrict__ Bw, long long sBz,
               float* __restrict__ C, long long cS1, long long cS2, int cZdiv,
               const float* __restrict__ bias,
               const float* __restrict__ Res, long long sRz,
               const float* __restrict__ scalep, float resScale,
               int Mi, int Nn, int Kk, int lda, int ldb, int ldc)
{
    __shared__ __align__(16) __bf16 As[2][TBM][LDSS];
    __shared__ __align__(16) __bf16 Bs[2][TBN][LDSS];

    const int z = blockIdx.z;
    A  += (long long)z * sAz;
    Bw += (long long)z * sBz;
    C  += (long long)(z / cZdiv) * cS1 + (long long)(z % cZdiv) * cS2;
    if (EPI == 2) Res += (long long)z * sRz;

    const int tid  = threadIdx.x;
    const int lane = tid & 31;
    const int wave = tid >> 5;
    const int wm   = wave & 3;          // 4 waves along M
    const int wn   = wave >> 2;         // 2 waves along N
    const int m0   = blockIdx.x * TBM;
    const int n0   = blockIdx.y * TBN;

    v8f acc[2][4];
    #pragma unroll
    for (int i = 0; i < 2; i++)
        #pragma unroll
        for (int j = 0; j < 4; j++) {
            v8f zv = {0.f,0.f,0.f,0.f,0.f,0.f,0.f,0.f};
            acc[i][j] = zv;
        }

    const float sc = scalep ? scalep[0] : 1.0f;
    const int ksteps = Kk / TBK;

    auto stageA = [&](int buf, int kt) {
        const int r = tid >> 1;
        const int c = (tid & 1) << 4;
        const float* src = A + (size_t)(m0 + r) * lda + kt * TBK + c;
        __bf16* dst = &As[buf][r][c];
        #pragma unroll
        for (int i = 0; i < 16; i++) dst[i] = (__bf16)src[i];
    };
    auto stageB = [&](int buf, int kt) {
        if constexpr (BT) {             // B memory layout [N,K] (bf16): pure copy
            const int r = tid >> 1;
            const int c = (tid & 1) << 4;
            const __bf16* src = Bw + (size_t)(n0 + r) * ldb + kt * TBK + c;
            uint4 d0 = {0u,0u,0u,0u}, d1 = {0u,0u,0u,0u};
            if (!NGUARD || (n0 + r) < Nn) {
                d0 = *reinterpret_cast<const uint4*>(src);
                d1 = *reinterpret_cast<const uint4*>(src + 8);
            }
            *reinterpret_cast<uint4*>(&Bs[buf][r][c])     = d0;
            *reinterpret_cast<uint4*>(&Bs[buf][r][c + 8]) = d1;
        } else {                        // B memory layout [K,N] (bf16); stage transposed
            const int k  = tid >> 3;
            const int nb = (tid & 7) << 4;
            const __bf16* src = Bw + (size_t)(kt * TBK + k) * ldb + n0 + nb;
            __bf16 tmp[16];
            if constexpr (NGUARD) {
                // Nn is a multiple of 16 -> chunk-granular guard keeps loads wide.
                if ((n0 + nb + 15) < Nn) {
                    *reinterpret_cast<uint4*>(tmp)     = *reinterpret_cast<const uint4*>(src);
                    *reinterpret_cast<uint4*>(tmp + 8) = *reinterpret_cast<const uint4*>(src + 8);
                } else {
                    #pragma unroll
                    for (int i = 0; i < 16; i++) tmp[i] = (__bf16)0.f;
                }
            } else {
                *reinterpret_cast<uint4*>(tmp)     = *reinterpret_cast<const uint4*>(src);
                *reinterpret_cast<uint4*>(tmp + 8) = *reinterpret_cast<const uint4*>(src + 8);
            }
            #pragma unroll
            for (int i = 0; i < 16; i++) Bs[buf][nb + i][k] = tmp[i];
        }
    };

    stageA(0, 0); stageB(0, 0);
    __syncthreads();

    const int ml = lane & 15;
    const int kh = lane >> 4;

    for (int kt = 0; kt < ksteps; ++kt) {
        const int buf = kt & 1;
        if (kt + 1 < ksteps) { stageA(buf ^ 1, kt + 1); stageB(buf ^ 1, kt + 1); }
        if (kt + 2 < ksteps) {  // gfx1250 global_prefetch_b8 for k+2 tile
            __builtin_prefetch(A + (size_t)(m0 + (tid >> 1)) * lda + (kt + 2) * TBK, 0, 1);
            if constexpr (BT)
                __builtin_prefetch(Bw + (size_t)(n0 + (tid >> 1)) * ldb + (kt + 2) * TBK, 0, 1);
            else
                __builtin_prefetch(Bw + (size_t)((kt + 2) * TBK + (tid >> 3)) * ldb + n0, 0, 1);
        }

        // Fragment loads per ISA 7.12.2 (16-bit A 16x32; B 32x16 staged [n][k]).
        v16bf af[2], bfr[4];
        #pragma unroll
        for (int i = 0; i < 2; i++) {
            const __bf16* p = &As[buf][wm * 32 + i * 16 + ml][kh * 8];
            #pragma unroll
            for (int t = 0; t < 8; t++) { af[i][t] = p[t]; af[i][8 + t] = p[16 + t]; }
        }
        #pragma unroll
        for (int j = 0; j < 4; j++) {
            const __bf16* p = &Bs[buf][wn * 64 + j * 16 + ml][kh * 8];
            #pragma unroll
            for (int t = 0; t < 8; t++) { bfr[j][t] = p[t]; bfr[j][8 + t] = p[16 + t]; }
        }
        #pragma unroll
        for (int i = 0; i < 2; i++)
            #pragma unroll
            for (int j = 0; j < 4; j++)
                acc[i][j] = __builtin_amdgcn_wmma_f32_16x16x32_bf16(
                    false, af[i], false, bfr[j], (short)0, acc[i][j], false, false);
        __syncthreads();
    }

    // Store per C/D layout: lanes 0-15 -> M=r, lanes 16-31 -> M=8+r; N = lane&15.
    const int cn0 = n0 + wn * 64 + ml;
    const int rb  = m0 + wm * 32 + kh * 8;
    #pragma unroll
    for (int j = 0; j < 4; j++) {
        const int n = cn0 + j * 16;
        if (NGUARD && n >= Nn) continue;
        const float bv = bias ? bias[n] : 0.f;
        #pragma unroll
        for (int i = 0; i < 2; i++) {
            #pragma unroll
            for (int r = 0; r < 8; r++) {
                const int m = rb + i * 16 + r;
                if (NGUARD && m >= Mi) continue;
                float v = acc[i][j][r] * sc + bv;
                if (EPI == 1) v = silu_f(v);
                if (EPI == 2) v = Res[(size_t)m * ldc + n] + resScale * v;
                C[(size_t)m * ldc + n] = v;
            }
        }
    }
}

// ------------------------------ LayerNorm ----------------------------------
__global__ void ln_kernel(const float* __restrict__ x, const float* __restrict__ g,
                          const float* __restrict__ b, float* __restrict__ out, int rows)
{
    const int row  = (blockIdx.x * blockDim.x + threadIdx.x) >> 5;
    const int lane = threadIdx.x & 31;
    if (row >= rows) return;
    const float* xr = x + (size_t)row * DMOD;
    float4 v[4]; float s = 0.f, s2 = 0.f;
    #pragma unroll
    for (int u = 0; u < 4; u++) {
        v[u] = *reinterpret_cast<const float4*>(xr + u * 128 + lane * 4);
        s  += v[u].x + v[u].y + v[u].z + v[u].w;
        s2 += v[u].x*v[u].x + v[u].y*v[u].y + v[u].z*v[u].z + v[u].w*v[u].w;
    }
    s = wred_sum(s); s2 = wred_sum(s2);
    const float mean = s * (1.f / DMOD);
    const float var  = s2 * (1.f / DMOD) - mean * mean;
    const float rs   = rsqrtf(var + 1e-5f);
    float* orow = out + (size_t)row * DMOD;
    #pragma unroll
    for (int u = 0; u < 4; u++) {
        const int c = u * 128 + lane * 4;
        float4 gg = *reinterpret_cast<const float4*>(g + c);
        float4 bb = *reinterpret_cast<const float4*>(b + c);
        float4 o;
        o.x = (v[u].x - mean) * rs * gg.x + bb.x;
        o.y = (v[u].y - mean) * rs * gg.y + bb.y;
        o.z = (v[u].z - mean) * rs * gg.z + bb.z;
        o.w = (v[u].w - mean) * rs * gg.w + bb.w;
        *reinterpret_cast<float4*>(orow + c) = o;
    }
}

// ---------------- qkv un-interleave + l2 normalize (q,k) -------------------
// qkv [tok,1536] with col = h*192 + d*3 + c.  Q -> fp32 (A of QK^T GEMM);
// K, V -> bf16 (B operands of the QK^T / attnV GEMMs).
__global__ void qkv_norm_kernel(const float* __restrict__ qkv,
                                float* __restrict__ Qn, __bf16* __restrict__ Kn,
                                __bf16* __restrict__ Vv)
{
    const int gw   = (blockIdx.x * blockDim.x + threadIdx.x) >> 5;
    const int lane = threadIdx.x & 31;
    const int t = gw >> 3;
    const int h = gw & 7;
    if (t >= MTOK) return;
    const float* base = qkv + (size_t)t * (3 * HHEADS * DHEAD) + h * (3 * DHEAD);
    float q[2], k[2], v[2];
    #pragma unroll
    for (int e = 0; e < 2; e++) {
        const int d = lane * 2 + e;
        q[e] = base[d * 3 + 0]; k[e] = base[d * 3 + 1]; v[e] = base[d * 3 + 2];
    }
    float sq = wred_sum(q[0]*q[0] + q[1]*q[1]);
    float sk = wred_sum(k[0]*k[0] + k[1]*k[1]);
    const float iq = 1.f / fmaxf(sqrtf(sq), 1e-12f);
    const float ik = 1.f / fmaxf(sqrtf(sk), 1e-12f);
    const int b = t >> 9, n = t & (NSEQ - 1);
    const size_t o = (((size_t)b * HHEADS + h) * NSEQ + n) * DHEAD;
    #pragma unroll
    for (int e = 0; e < 2; e++) {
        const int d = lane * 2 + e;
        Qn[o + d] = q[e] * iq;
        Kn[o + d] = (__bf16)(k[e] * ik);
        Vv[o + d] = (__bf16)v[e];
    }
}

// ------------- talking-heads mix + mask + softmax, in place on S -----------
// One block per (b,i): owns all 8 mixed heads for row i -> in-place is safe.
__global__ __launch_bounds__(256)
void th_softmax_kernel(float* __restrict__ S, const float* __restrict__ tw,
                       const float* __restrict__ tb, const unsigned char* __restrict__ pad)
{
    __shared__ float Ss[HHEADS][NSEQ];   // 16KB
    __shared__ float twl[HHEADS * HHEADS];
    __shared__ float tbl[HHEADS];
    __shared__ float red[8];
    const int tid = threadIdx.x;
    const int bb  = blockIdx.x >> 9;
    const int i   = blockIdx.x & (NSEQ - 1);
    if (tid < HHEADS * HHEADS) twl[tid] = tw[tid];
    if (tid < HHEADS)          tbl[tid] = tb[tid];
    #pragma unroll
    for (int h = 0; h < HHEADS; h++) {
        const size_t base = (((size_t)bb * HHEADS + h) * NSEQ + i) * NSEQ;
        Ss[h][tid]       = S[base + tid];
        Ss[h][tid + 256] = S[base + tid + 256];
    }
    __syncthreads();
    const bool vi  = !pad[bb * NSEQ + i];
    const int lane = tid & 31, w = tid >> 5;

    for (int g = 0; g < HHEADS; ++g) {
        float val[2];
        #pragma unroll
        for (int u = 0; u < 2; u++) {
            const int j = tid + u * 256;
            float a = tbl[g];
            #pragma unroll
            for (int h = 0; h < HHEADS; h++) a += twl[g * HHEADS + h] * Ss[h][j];
            const bool vj = !pad[bb * NSEQ + j];
            val[u] = (vi && vj) ? a : -3.402823466e38f;
        }
        float mx = wred_max(fmaxf(val[0], val[1]));
        if (lane == 0) red[w] = mx;
        __syncthreads();
        mx = red[0];
        #pragma unroll
        for (int kk = 1; kk < 8; kk++) mx = fmaxf(mx, red[kk]);
        __syncthreads();
        const float e0 = expf(val[0] - mx), e1 = expf(val[1] - mx);
        float sm = wred_sum(e0 + e1);
        if (lane == 0) red[w] = sm;
        __syncthreads();
        sm = 0.f;
        #pragma unroll
        for (int kk = 0; kk < 8; kk++) sm += red[kk];
        __syncthreads();
        const float inv = 1.f / sm;
        const size_t ob = (((size_t)bb * HHEADS + g) * NSEQ + i) * NSEQ;
        S[ob + tid]       = e0 * inv;
        S[ob + tid + 256] = e1 * inv;
    }
}

// ------------------------------ GLU + pad mask -----------------------------
__global__ void glu_kernel(const float* __restrict__ raw,
                           const unsigned char* __restrict__ pad, float* __restrict__ out)
{
    const int e  = blockIdx.x * blockDim.x + threadIdx.x;  // over MTOK*128 float4's
    const int t  = e >> 7;
    const int d4 = (e & 127) * 4;
    float4 a = *reinterpret_cast<const float4*>(raw + (size_t)t * 1024 + d4);
    float4 g = *reinterpret_cast<const float4*>(raw + (size_t)t * 1024 + 512 + d4);
    const float m = pad[t] ? 0.f : 1.f;
    float4 o;
    o.x = m * a.x * sigm_f(g.x); o.y = m * a.y * sigm_f(g.y);
    o.z = m * a.z * sigm_f(g.z); o.w = m * a.w * sigm_f(g.w);
    *reinterpret_cast<float4*>(out + (size_t)t * DMOD + d4) = o;
}

// ------------- depthwise conv (K=31) + BN(inference) + SiLU ----------------
__global__ void dwconv_kernel(const float* __restrict__ gin, const float* __restrict__ w,
                              const float* __restrict__ wb, const float* __restrict__ bng,
                              const float* __restrict__ bnb, float* __restrict__ out)
{
    const int e  = blockIdx.x * blockDim.x + threadIdx.x;
    const int t  = e >> 7;
    const int d4 = (e & 127) * 4;
    const int b = t >> 9, n = t & (NSEQ - 1);
    float4 acc = *reinterpret_cast<const float4*>(wb + d4);
    #pragma unroll
    for (int k = 0; k < KCONV; k++) {
        const int ns = n + k - (KCONV - 1) / 2;
        if (ns < 0 || ns >= NSEQ) continue;
        float4 gv = *reinterpret_cast<const float4*>(gin + (size_t)(b * NSEQ + ns) * DMOD + d4);
        float4 wv = *reinterpret_cast<const float4*>(w + (size_t)k * DMOD + d4);
        acc.x += gv.x * wv.x; acc.y += gv.y * wv.y;
        acc.z += gv.z * wv.z; acc.w += gv.w * wv.w;
    }
    const float bns = rsqrtf(1.f + 1e-5f);
    float4 gg = *reinterpret_cast<const float4*>(bng + d4);
    float4 bb = *reinterpret_cast<const float4*>(bnb + d4);
    float4 o;
    o.x = silu_f(acc.x * gg.x * bns + bb.x);
    o.y = silu_f(acc.y * gg.y * bns + bb.y);
    o.z = silu_f(acc.z * gg.z * bns + bb.z);
    o.w = silu_f(acc.w * gg.w * bns + bb.w);
    *reinterpret_cast<float4*>(out + (size_t)t * DMOD + d4) = o;
}

// ---------------------- softmax over V=128 (in place) ----------------------
__global__ void softmax_v_kernel(float* __restrict__ P)
{
    const int row  = (blockIdx.x * blockDim.x + threadIdx.x) >> 5;
    const int lane = threadIdx.x & 31;
    float* pr = P + (size_t)row * VVOC;
    float v[4];
    #pragma unroll
    for (int u = 0; u < 4; u++) v[u] = pr[lane + u * 32];
    float mx = wred_max(fmaxf(fmaxf(v[0], v[1]), fmaxf(v[2], v[3])));
    float s = 0.f;
    #pragma unroll
    for (int u = 0; u < 4; u++) { v[u] = expf(v[u] - mx); s += v[u]; }
    s = wred_sum(s);
    const float inv = 1.f / s;
    #pragma unroll
    for (int u = 0; u < 4; u++) pr[lane + u * 32] = v[u] * inv;
}

// ------------------------------ orchestration ------------------------------
extern "C" void kernel_launch(void* const* d_in, const int* in_sizes, int n_in,
                              void* d_out, int out_size, void* d_ws, size_t ws_size,
                              hipStream_t stream)
{
    (void)in_sizes; (void)n_in; (void)out_size; (void)ws_size;
    const float* x      = (const float*)d_in[0];
    const unsigned char* pad = (const unsigned char*)d_in[1];
    const float* ln_g   = (const float*)d_in[2];
    const float* ln_b   = (const float*)d_in[3];
    const float* ff1_w1 = (const float*)d_in[4];
    const float* ff1_b1 = (const float*)d_in[5];
    const float* ff1_w2 = (const float*)d_in[6];
    const float* ff1_b2 = (const float*)d_in[7];
    const float* ff2_w1 = (const float*)d_in[8];
    const float* ff2_b1 = (const float*)d_in[9];
    const float* ff2_w2 = (const float*)d_in[10];
    const float* ff2_b2 = (const float*)d_in[11];
    const float* qkv_w  = (const float*)d_in[12];
    const float* temp   = (const float*)d_in[13];
    const float* th_w   = (const float*)d_in[14];
    const float* th_b   = (const float*)d_in[15];
    const float* out_w  = (const float*)d_in[16];
    const float* pw1_w  = (const float*)d_in[17];
    const float* pw1_b  = (const float*)d_in[18];
    const float* dw_w   = (const float*)d_in[19];
    const float* dw_b   = (const float*)d_in[20];
    const float* bn_g   = (const float*)d_in[21];
    const float* bn_b   = (const float*)d_in[22];
    const float* pw2_w  = (const float*)d_in[23];
    const float* pw2_b  = (const float*)d_in[24];
    const float* dec_w  = (const float*)d_in[25];
    const float* dec_b  = (const float*)d_in[26];
    const float* rep_w  = (const float*)d_in[27];
    const float* rep_b  = (const float*)d_in[28];
    float* out = (float*)d_out;

    // Workspace carve-up (byte-based).
    char* wsb = (char*)d_ws;
    size_t off = 0;
    auto carve = [&](size_t bytes) { void* p = wsb + off; off += (bytes + 255) & ~(size_t)255; return p; };
    float*  resb = (float*)carve((size_t)MTOK * DMOD * 4);
    float*  lnb  = (float*)carve((size_t)MTOK * DMOD * 4);
    float*  big  = (float*)carve((size_t)MTOK * FFDIM * 4);
    float*  Qn   = (float*)carve((size_t)MTOK * DMOD * 4);
    __bf16* Knb  = (__bf16*)carve((size_t)MTOK * DMOD * 2);
    __bf16* Vvb  = (__bf16*)carve((size_t)MTOK * DMOD * 2);
    float*  S    = (float*)carve((size_t)BBAT * HHEADS * NSEQ * NSEQ * 4);  // 64MB
    float*  Ob   = (float*)carve((size_t)MTOK * DMOD * 4);
    float*  Gb   = (float*)carve((size_t)MTOK * DMOD * 4);
    float*  Cb   = (float*)carve((size_t)MTOK * DMOD * 4);
    float*  Pb   = (float*)carve((size_t)MTOK * VVOC * 4);
    float*  Hb   = (float*)carve((size_t)MTOK * DMOD * 4);
    // bf16 transposed weight mirrors [N,K] (converted once per launch).
    __bf16* wf1a = (__bf16*)carve((size_t)LNUM * DMOD * FFDIM * 2);
    __bf16* wf1b = (__bf16*)carve((size_t)LNUM * FFDIM * DMOD * 2);
    __bf16* wf2a = (__bf16*)carve((size_t)LNUM * DMOD * FFDIM * 2);
    __bf16* wf2b = (__bf16*)carve((size_t)LNUM * FFDIM * DMOD * 2);
    __bf16* wqkv = (__bf16*)carve((size_t)LNUM * DMOD * 1536 * 2);
    __bf16* wout = (__bf16*)carve((size_t)LNUM * DMOD * DMOD * 2);
    __bf16* wpw1 = (__bf16*)carve((size_t)LNUM * DMOD * 1024 * 2);
    __bf16* wpw2 = (__bf16*)carve((size_t)LNUM * DMOD * DMOD * 2);
    __bf16* wdec = (__bf16*)carve((size_t)LNUM * DMOD * VVOC * 2);
    __bf16* wrep = (__bf16*)carve((size_t)LNUM * VVOC * DMOD * 2);

    const dim3 blk(256);

    // One-time fp32 [L,K,N] -> bf16 [L,N,K] weight transposes.
    auto cvtT = [&](const float* s, __bf16* d, int K, int N) {
        cvtT_kernel<<<dim3(K / 32, N / 32, LNUM), blk, 0, stream>>>(s, d, K, N);
    };
    cvtT(ff1_w1, wf1a, DMOD,  FFDIM);
    cvtT(ff1_w2, wf1b, FFDIM, DMOD);
    cvtT(ff2_w1, wf2a, DMOD,  FFDIM);
    cvtT(ff2_w2, wf2b, FFDIM, DMOD);
    cvtT(qkv_w,  wqkv, DMOD,  1536);
    cvtT(out_w,  wout, DMOD,  DMOD);
    cvtT(pw1_w,  wpw1, DMOD,  1024);
    cvtT(pw2_w,  wpw2, DMOD,  DMOD);
    cvtT(dec_w,  wdec, DMOD,  VVOC);
    cvtT(rep_w,  wrep, VVOC,  DMOD);

    for (int l = 0; l < LNUM; l++) {
        const float* hin = (l == 0) ? x : Hb;
        const float* lg  = ln_g + (size_t)l * 5 * DMOD;
        const float* lb  = ln_b + (size_t)l * 5 * DMOD;

        // ---- FF1 half-step: res = h + 0.5*(silu(ln(h)@w1+b1)@w2+b2)
        ln_kernel<<<MTOK / 8, blk, 0, stream>>>(hin, lg, lb, lnb, MTOK);
        gemm_wmma<1, true, false><<<dim3(MTOK / 128, FFDIM / 128, 1), blk, 0, stream>>>(
            lnb, 0, wf1a + (size_t)l * DMOD * FFDIM, 0, big, 0, 0, 1,
            ff1_b1 + (size_t)l * FFDIM, nullptr, 0, nullptr, 1.f,
            MTOK, FFDIM, DMOD, DMOD, DMOD, FFDIM);
        gemm_wmma<2, true, false><<<dim3(MTOK / 128, DMOD / 128, 1), blk, 0, stream>>>(
            big, 0, wf1b + (size_t)l * FFDIM * DMOD, 0, resb, 0, 0, 1,
            ff1_b2 + (size_t)l * DMOD, hin, 0, nullptr, 0.5f,
            MTOK, DMOD, FFDIM, FFDIM, FFDIM, DMOD);

        // ---- Cosine attention with talking heads
        ln_kernel<<<MTOK / 8, blk, 0, stream>>>(resb, lg + DMOD, lb + DMOD, lnb, MTOK);
        gemm_wmma<0, true, false><<<dim3(MTOK / 128, 1536 / 128, 1), blk, 0, stream>>>(
            lnb, 0, wqkv + (size_t)l * DMOD * 1536, 0, big, 0, 0, 1,
            nullptr, nullptr, 0, nullptr, 1.f,
            MTOK, 1536, DMOD, DMOD, DMOD, 1536);
        qkv_norm_kernel<<<MTOK * HHEADS / 8, blk, 0, stream>>>(big, Qn, Knb, Vvb);
        // dots[b,h,i,j] = (Qn Kn^T) * temp   (B transposed layout, K=64)
        gemm_wmma<0, true, false><<<dim3(NSEQ / 128, NSEQ / 128, BBAT * HHEADS), blk, 0, stream>>>(
            Qn, (long long)NSEQ * DHEAD, Knb, (long long)NSEQ * DHEAD,
            S, (long long)NSEQ * NSEQ, 0, 1,
            nullptr, nullptr, 0, temp + l, 1.f,
            NSEQ, NSEQ, DHEAD, DHEAD, DHEAD, NSEQ);
        th_softmax_kernel<<<BBAT * NSEQ, blk, 0, stream>>>(
            S, th_w + (size_t)l * HHEADS * HHEADS, th_b + (size_t)l * HHEADS, pad);
        // O[b,n,h*64+d] = P @ V   (split C offsets: z=(b*H+h)); N=64 -> guarded
        gemm_wmma<0, false, true><<<dim3(NSEQ / 128, 1, BBAT * HHEADS), blk, 0, stream>>>(
            S, (long long)NSEQ * NSEQ, Vvb, (long long)NSEQ * DHEAD,
            Ob, (long long)NSEQ * DMOD, (long long)DHEAD, HHEADS,
            nullptr, nullptr, 0, nullptr, 1.f,
            NSEQ, DHEAD, NSEQ, NSEQ, DHEAD, DMOD);
        gemm_wmma<2, true, false><<<dim3(MTOK / 128, DMOD / 128, 1), blk, 0, stream>>>(
            Ob, 0, wout + (size_t)l * DMOD * DMOD, 0, resb, 0, 0, 1,
            nullptr, resb, 0, nullptr, 1.f,
            MTOK, DMOD, DMOD, DMOD, DMOD, DMOD);

        // ---- Conv module
        ln_kernel<<<MTOK / 8, blk, 0, stream>>>(resb, lg + 2 * DMOD, lb + 2 * DMOD, lnb, MTOK);
        gemm_wmma<0, true, false><<<dim3(MTOK / 128, 1024 / 128, 1), blk, 0, stream>>>(
            lnb, 0, wpw1 + (size_t)l * DMOD * 1024, 0, big, 0, 0, 1,
            pw1_b + (size_t)l * 1024, nullptr, 0, nullptr, 1.f,
            MTOK, 1024, DMOD, DMOD, DMOD, 1024);
        glu_kernel<<<MTOK * 128 / 256, blk, 0, stream>>>(big, pad, Gb);
        dwconv_kernel<<<MTOK * 128 / 256, blk, 0, stream>>>(
            Gb, dw_w + (size_t)l * KCONV * DMOD, dw_b + (size_t)l * DMOD,
            bn_g + (size_t)l * DMOD, bn_b + (size_t)l * DMOD, Cb);
        gemm_wmma<2, true, false><<<dim3(MTOK / 128, DMOD / 128, 1), blk, 0, stream>>>(
            Cb, 0, wpw2 + (size_t)l * DMOD * DMOD, 0, resb, 0, 0, 1,
            pw2_b + (size_t)l * DMOD, resb, 0, nullptr, 1.f,
            MTOK, DMOD, DMOD, DMOD, DMOD, DMOD);

        // ---- FF2 half-step
        ln_kernel<<<MTOK / 8, blk, 0, stream>>>(resb, lg + 3 * DMOD, lb + 3 * DMOD, lnb, MTOK);
        gemm_wmma<1, true, false><<<dim3(MTOK / 128, FFDIM / 128, 1), blk, 0, stream>>>(
            lnb, 0, wf2a + (size_t)l * DMOD * FFDIM, 0, big, 0, 0, 1,
            ff2_b1 + (size_t)l * FFDIM, nullptr, 0, nullptr, 1.f,
            MTOK, FFDIM, DMOD, DMOD, DMOD, FFDIM);
        gemm_wmma<2, true, false><<<dim3(MTOK / 128, DMOD / 128, 1), blk, 0, stream>>>(
            big, 0, wf2b + (size_t)l * FFDIM * DMOD, 0, resb, 0, 0, 1,
            ff2_b2 + (size_t)l * DMOD, resb, 0, nullptr, 0.5f,
            MTOK, DMOD, FFDIM, FFDIM, FFDIM, DMOD);

        // ---- final LN (last layer writes straight to d_out)
        float* hout = (l == LNUM - 1) ? out : Hb;
        ln_kernel<<<MTOK / 8, blk, 0, stream>>>(resb, lg + 4 * DMOD, lb + 4 * DMOD, hout, MTOK);

        // ---- self-conditioning between layers
        if (l < LNUM - 1) {
            gemm_wmma<0, true, false><<<dim3(MTOK / 128, 1, 1), blk, 0, stream>>>(
                Hb, 0, wdec + (size_t)l * DMOD * VVOC, 0, Pb, 0, 0, 1,
                dec_b + (size_t)l * VVOC, nullptr, 0, nullptr, 1.f,
                MTOK, VVOC, DMOD, DMOD, DMOD, VVOC);
            softmax_v_kernel<<<MTOK / 8, blk, 0, stream>>>(Pb);
            gemm_wmma<2, true, false><<<dim3(MTOK / 128, DMOD / 128, 1), blk, 0, stream>>>(
                Pb, 0, wrep + (size_t)l * VVOC * DMOD, 0, Hb, 0, 0, 1,
                rep_b + (size_t)l * DMOD, Hb, 0, nullptr, 1.f,
                MTOK, DMOD, VVOC, VVOC, VVOC, DMOD);
        }
    }
}
